// RNFSEncoder_33011118637764
// MI455X (gfx1250) — compile-verified, matrix operand order checked by reading, not run on hardware
//
#include <hip/hip_runtime.h>
#include <hip/hip_bf16.h>

typedef __attribute__((ext_vector_type(16))) _Float16 v16h;
typedef __attribute__((ext_vector_type(8)))  _Float16 v8h;
typedef __attribute__((ext_vector_type(8)))  float    v8f;
typedef __attribute__((ext_vector_type(4)))  int      v4i;

// Problem constants (reference: B=32, L=64, C=128, D=512, O=256, L2=4096)
#define RB   32
#define RL   64
#define RC   128
#define RD   512
#define RO   256
#define RL2  4096
#define R2C  256

#if defined(__HIP_DEVICE_COMPILE__) && \
    __has_builtin(__builtin_amdgcn_global_load_async_to_lds_b128) && \
    __has_builtin(__builtin_amdgcn_s_wait_asynccnt)
#define USE_ASYNC_LDS 1
typedef __attribute__((address_space(1))) v4i g_v4i;   // global addrspace
typedef __attribute__((address_space(3))) v4i l_v4i;   // LDS addrspace
#else
#define USE_ASYNC_LDS 0
#endif

// ---------------------------------------------------------------------------
// f32 -> f16 conversion (contiguous)
// ---------------------------------------------------------------------------
__global__ __launch_bounds__(256) void cvt_f32_f16(const float* __restrict__ in,
                                                   _Float16* __restrict__ out,
                                                   int n) {
  int i = blockIdx.x * blockDim.x + threadIdx.x;
  if (i < n) out[i] = (_Float16)in[i];
}

// ---------------------------------------------------------------------------
// f32 [K][N] -> f16 transposed [N][K] (one-shot; weights stay L2-resident)
// ---------------------------------------------------------------------------
__global__ __launch_bounds__(256) void cvt_transpose_f16(
    const float* __restrict__ in,   // [K][N]
    _Float16* __restrict__ out,     // [N][K]
    int K, int N) {
  int idx = blockIdx.x * blockDim.x + threadIdx.x;
  if (idx < K * N) {
    int n = idx / K;
    int k = idx - n * K;
    out[idx] = (_Float16)in[k * N + n];
  }
}

// ---------------------------------------------------------------------------
// Fused pair-MLP: workgroup (b, i) computes rows p = i*64 + j, j=0..63:
//   pairs[p] = concat(x[b,i], x[b,j])   (built on the fly from LDS copy of x[b])
//   h = relu(pairs @ w1 + b1)           (64x512 f16, kept in LDS)
//   y = h @ w2 + b2                     (64x256 f32 -> workspace)
// 8 waves / 256 threads. B fragments come straight from pre-transposed
// weights in global (L2/WGP$-resident) as contiguous b128 loads -> the K
// loops contain no barriers at all.
// ---------------------------------------------------------------------------
__global__ __launch_bounds__(256) void mlp_pairs_kernel(
    const _Float16* __restrict__ xh,    // [B][64][128] f16
    const _Float16* __restrict__ w1T,   // [512][256]   f16  (N-major)
    const float*    __restrict__ b1,    // [512]
    const _Float16* __restrict__ w2T,   // [256][512]   f16  (N-major)
    const float*    __restrict__ b2,    // [256]
    float*          __restrict__ y)     // [B][4096][256] f32
{
  __shared__ __align__(16) _Float16 xs[RL * RC];   // 16 KB: x[b]
  __shared__ __align__(16) _Float16 hs[RL * RD];   // 64 KB: h tile

  const int b    = blockIdx.x >> 6;
  const int irow = blockIdx.x & 63;
  const int tid  = threadIdx.x;
  const int wave = tid >> 5;
  const int lane = tid & 31;
  const int r    = lane & 15;          // row/col within 16-wide tile
  const int kb   = (lane >> 4) * 8;    // WMMA K sub-block base (0 or 8)

  // ---- bring x[b] into LDS (8192 halfs = 1024 x 16B chunks) ----
  {
    const _Float16* xsrc = xh + (size_t)b * (RL * RC);
#if USE_ASYNC_LDS
    #pragma unroll
    for (int t = 0; t < 4; ++t) {
      int c = tid + t * 256;   // 16B chunk index
      __builtin_amdgcn_global_load_async_to_lds_b128(
          (g_v4i*)(xsrc + c * 8), (l_v4i*)(xs + c * 8), 0, 0);
    }
    __builtin_amdgcn_s_wait_asynccnt(0);
#else
    const v8h* src = (const v8h*)xsrc;
    v8h* dst = (v8h*)xs;
    #pragma unroll
    for (int t = 0; t < 4; ++t) dst[tid + t * 256] = src[tid + t * 256];
#endif
  }
  __syncthreads();

  // =======================================================================
  // GEMM1: [64 x 256] x [256 x 512] -> h, each wave owns 64 N-columns
  // =======================================================================
  const int nb1 = wave * 64;
  v8f acc1[4][4];
  #pragma unroll
  for (int nt = 0; nt < 4; ++nt) {
    float bias = b1[nb1 + nt * 16 + r];
    #pragma unroll
    for (int mt = 0; mt < 4; ++mt)
      #pragma unroll
      for (int v = 0; v < 8; ++v) acc1[mt][nt][v] = bias;
  }

  for (int kk = 0; kk < R2C / 32; ++kk) {
    const int K0 = kk * 32 + kb;

    // A fragments from LDS: pairs row = concat(x[b,irow], x[b,j])
    v16h afrag[4];
    #pragma unroll
    for (int mt = 0; mt < 4; ++mt) {
      int row = mt * 16 + r;  // j index
      const _Float16* rowp = (K0 < RC) ? (xs + irow * RC) : (xs + row * RC - RC);
      v8h lo = *(const v8h*)(rowp + K0);
      v8h hi = *(const v8h*)(rowp + K0 + 16);
      afrag[mt] = __builtin_shufflevector(lo, hi, 0,1,2,3,4,5,6,7,8,9,10,11,12,13,14,15);
    }

    // B fragments straight from pre-transposed w1 (global, cache-resident)
    #pragma unroll
    for (int nt = 0; nt < 4; ++nt) {
      const _Float16* colp = w1T + (size_t)(nb1 + nt * 16 + r) * R2C + K0;
      v8h lo = *(const v8h*)(colp);
      v8h hi = *(const v8h*)(colp + 16);
      if (kk + 1 < R2C / 32) __builtin_prefetch(colp + 32, 0, 1);
      v16h bfrag = __builtin_shufflevector(lo, hi, 0,1,2,3,4,5,6,7,8,9,10,11,12,13,14,15);
      #pragma unroll
      for (int mt = 0; mt < 4; ++mt)
        acc1[mt][nt] = __builtin_amdgcn_wmma_f32_16x16x32_f16(
            false, afrag[mt], false, bfrag, (short)0, acc1[mt][nt], false, false);
    }
  }

  // relu + store h tile to LDS as f16
  {
    const int mrow = (lane >> 4) * 8;
    #pragma unroll
    for (int mt = 0; mt < 4; ++mt)
      #pragma unroll
      for (int nt = 0; nt < 4; ++nt) {
        int n = nb1 + nt * 16 + r;
        #pragma unroll
        for (int v = 0; v < 8; ++v) {
          float hv = acc1[mt][nt][v];
          hs[(mt * 16 + mrow + v) * RD + n] = (_Float16)(hv > 0.f ? hv : 0.f);
        }
      }
  }
  __syncthreads();

  // =======================================================================
  // GEMM2: [64 x 512] x [512 x 256] -> y, each wave owns 32 N-columns
  // =======================================================================
  const int nb2 = wave * 32;
  v8f acc2[4][2];
  #pragma unroll
  for (int nt = 0; nt < 2; ++nt) {
    float bias = b2[nb2 + nt * 16 + r];
    #pragma unroll
    for (int mt = 0; mt < 4; ++mt)
      #pragma unroll
      for (int v = 0; v < 8; ++v) acc2[mt][nt][v] = bias;
  }

  for (int kk = 0; kk < RD / 32; ++kk) {
    const int K0 = kk * 32 + kb;

    v16h afrag[4];
    #pragma unroll
    for (int mt = 0; mt < 4; ++mt) {
      const _Float16* rowp = hs + (mt * 16 + r) * RD + K0;
      v8h lo = *(const v8h*)(rowp);
      v8h hi = *(const v8h*)(rowp + 16);
      afrag[mt] = __builtin_shufflevector(lo, hi, 0,1,2,3,4,5,6,7,8,9,10,11,12,13,14,15);
    }

    #pragma unroll
    for (int nt = 0; nt < 2; ++nt) {
      const _Float16* colp = w2T + (size_t)(nb2 + nt * 16 + r) * RD + K0;
      v8h lo = *(const v8h*)(colp);
      v8h hi = *(const v8h*)(colp + 16);
      if (kk + 1 < RD / 32) __builtin_prefetch(colp + 32, 0, 1);
      v16h bfrag = __builtin_shufflevector(lo, hi, 0,1,2,3,4,5,6,7,8,9,10,11,12,13,14,15);
      #pragma unroll
      for (int mt = 0; mt < 4; ++mt)
        acc2[mt][nt] = __builtin_amdgcn_wmma_f32_16x16x32_f16(
            false, afrag[mt], false, bfrag, (short)0, acc2[mt][nt], false, false);
    }
  }

  // write y tile [64 x 256] f32
  {
    float* ybase = y + ((size_t)b * RL2 + (size_t)irow * 64) * RO;
    const int mrow = (lane >> 4) * 8;
    #pragma unroll
    for (int mt = 0; mt < 4; ++mt)
      #pragma unroll
      for (int nt = 0; nt < 2; ++nt) {
        int n = nb2 + nt * 16 + r;
        #pragma unroll
        for (int v = 0; v < 8; ++v)
          ybase[(size_t)(mt * 16 + mrow + v) * RO + n] = acc2[mt][nt][v];
      }
  }
}

// ---------------------------------------------------------------------------
// FSPool: per (b,o) sort y[b,:,o] ascending (bitonic in LDS), dot pool_w[:,o]
// ---------------------------------------------------------------------------
__global__ __launch_bounds__(256) void fspool_kernel(
    const float* __restrict__ y,       // [B][4096][256]
    const float* __restrict__ pool_w,  // [4096][256]
    float*       __restrict__ out)     // [B][256]
{
  __shared__ float s[RL2];   // 16 KB
  __shared__ float red[8];

  const int b   = blockIdx.x >> 8;
  const int o   = blockIdx.x & 255;
  const int tid = threadIdx.x;

  const float* col = y + (size_t)b * RL2 * RO + o;
  #pragma unroll
  for (int t = 0; t < RL2 / 256; ++t) {
    int l = tid + t * 256;
    s[l] = col[(size_t)l * RO];
  }
  __syncthreads();

  // bitonic sort, ascending
  for (int k = 2; k <= RL2; k <<= 1) {
    for (int j = k >> 1; j > 0; j >>= 1) {
      #pragma unroll
      for (int t = 0; t < RL2 / 256; ++t) {
        int idx = tid + t * 256;
        int partner = idx ^ j;
        if (partner > idx) {
          bool up = (idx & k) == 0;
          float a = s[idx], c = s[partner];
          if ((a > c) == up) { s[idx] = c; s[partner] = a; }
        }
      }
      __syncthreads();
    }
  }

  // dot with pool_w column
  float acc = 0.f;
  #pragma unroll
  for (int t = 0; t < RL2 / 256; ++t) {
    int l = tid + t * 256;
    acc += s[l] * pool_w[(size_t)l * RO + o];
  }
  #pragma unroll
  for (int off = 16; off > 0; off >>= 1) acc += __shfl_down(acc, off, 32);
  if ((tid & 31) == 0) red[tid >> 5] = acc;
  __syncthreads();
  if (tid == 0) {
    float t = 0.f;
    #pragma unroll
    for (int w = 0; w < 8; ++w) t += red[w];
    out[b * RO + o] = t;
  }
}

// ---------------------------------------------------------------------------
extern "C" void kernel_launch(void* const* d_in, const int* in_sizes, int n_in,
                              void* d_out, int out_size, void* d_ws, size_t ws_size,
                              hipStream_t stream) {
  const float* x      = (const float*)d_in[0];  // [32][64][128]
  const float* w1     = (const float*)d_in[1];  // [256][512]
  const float* b1     = (const float*)d_in[2];  // [512]
  const float* w2     = (const float*)d_in[3];  // [512][256]
  const float* b2     = (const float*)d_in[4];  // [256]
  const float* pool_w = (const float*)d_in[5];  // [4096][256]
  float* out = (float*)d_out;                   // [32][256]

  char* wsb = (char*)d_ws;
  _Float16* xh  = (_Float16*)(wsb);                       // 262144 halfs
  _Float16* w1T = (_Float16*)(wsb + 524288);              // 131072 halfs [512][256]
  _Float16* w2T = (_Float16*)(wsb + 524288 + 262144);     // 131072 halfs [256][512]
  float*    y   = (float*)   (wsb + 1048576);             // 128 MB

  cvt_f32_f16<<<(262144 + 255) / 256, 256, 0, stream>>>(x, xh, 262144);
  cvt_transpose_f16<<<(131072 + 255) / 256, 256, 0, stream>>>(w1, w1T, R2C, RD);
  cvt_transpose_f16<<<(131072 + 255) / 256, 256, 0, stream>>>(w2, w2T, RD, RO);

  mlp_pairs_kernel<<<RB * RL, 256, 0, stream>>>(xh, w1T, b1, w2T, b2, y);

  fspool_kernel<<<RB * RO, 256, 0, stream>>>(y, pool_w, out);
}
